// VisualCortex_22797686408202
// MI455X (gfx1250) — matrix operand order repaired and checked by reading, not compiled, not run on hardware
//
#include <hip/hip_runtime.h>

typedef __attribute__((ext_vector_type(16))) _Float16 v16h;
typedef __attribute__((ext_vector_type(8)))  float    v8f;
typedef unsigned int u32x4 __attribute__((ext_vector_type(4)));
typedef int          i32x4 __attribute__((ext_vector_type(4)));
typedef int          i32x8 __attribute__((ext_vector_type(8)));

#define INPUT_DIM 784
#define HIDDEN    4096
#define BATCH     64
#define T_STEPS   30
#define DECAY_F   (1.0f - 1.0f / 100.0f)
#define THR_F     0.5f
#define SCALE_F   16.0f
#define K1P       800              // INPUT_DIM padded up to multiple of 32
#define KT1       (K1P / 32)       // 25 K-tiles for GEMM1
#define KT2       (HIDDEN / 32)    // 128 K-tiles for GEMM2
#define KSEG1     5                // GEMM1 K-split: 5 segments x 5 ktiles
#define KSEG2     8                // GEMM2 K-split: 8 segments x 16 ktiles
#define HID_ELEMS (BATCH * HIDDEN)     // 262144
#define IN_ELEMS  (BATCH * INPUT_DIM)  // 50176
#define FRAG_PER_KT 2048           // f16 per K-tile of A fragments (4 mt * 32 lanes * 16)
#define SH_ELEMS  (16 * FRAG_PER_KT)   // max K-seg slice = 32768 f16 = 64 KB LDS

// ---- WMMA fragment-order address helpers (per CDNA5 ISA 7.12.2) ----------
// A (MxK, f16): lane r (0-15) = row r, K chunks {0..7, 16..23};
//               lane r+16     = row r, K chunks {8..15, 24..31}.
__device__ __forceinline__ size_t a_frag_off(int m, int k) {
  int mt = m >> 4, r = m & 15, kt = k >> 5, kk = k & 31;
  int grp  = kk >> 3;                       // which 8-wide K chunk
  int lane = r + ((grp & 1) << 4);
  int pos  = (kk & 7) + ((grp >> 1) << 3);  // f16 slot within lane's 16
  return ((size_t)(kt * 4 + mt) * 32 + lane) * 16 + pos;  // 4 = BATCH/16 M-tiles
}

// ---- Tensor Data Mover: 1-D contiguous global -> LDS copy -----------------
// D# per ISA 08_async_tensor §8.3/§8.4: count=1, type=2("image"),
// data_size=1 (2 bytes), single row of `nelem` f16: tensor_dim0 = tile_dim0
// = nelem, tensor_dim1 = tile_dim1 = 1, tensor_dim0_stride = nelem.
__device__ __forceinline__ void tdm_load_1d(const void* gsrc, unsigned lds_off,
                                            unsigned nelem) {
  unsigned long long ga = (unsigned long long)gsrc;
  u32x4 g0 = { 1u,                                   // count=1, user desc
               lds_off,                              // lds_addr (bytes)
               (unsigned)(ga & 0xFFFFFFFFu),         // global_addr[31:0]
               (unsigned)((ga >> 32) & 0x01FFFFFFu) | (2u << 30) }; // addr hi | type=2
  i32x8 g1;
  g1[0] = (int)(1u << 16);                           // data_size=1 (2B), no flags
  g1[1] = (int)((nelem & 0xFFFFu) << 16);            // tensor_dim0[15:0] @ bits 63:48
  g1[2] = (int)((nelem >> 16) | (1u << 16));         // tensor_dim0 hi | tensor_dim1=1
  g1[3] = (int)((nelem & 0xFFFFu) << 16);            // tile_dim0 @ bits 127:112
  g1[4] = 1;                                         // tile_dim1=1, tile_dim2=0
  g1[5] = (int)nelem;                                // tensor_dim0_stride[31:0]
  g1[6] = 0;                                         // stride hi, dim1_stride lo
  g1[7] = 0;
  i32x4 gz = { 0, 0, 0, 0 };
#if __clang_major__ >= 23
  i32x8 gz8 = { 0, 0, 0, 0, 0, 0, 0, 0 };
  __builtin_amdgcn_tensor_load_to_lds(g0, g1, gz, gz, gz8, 0);
#else
  __builtin_amdgcn_tensor_load_to_lds(g0, g1, gz, gz, 0);
#endif
}

// ---- kernels --------------------------------------------------------------

__global__ void snn_zero_f32(float* __restrict__ p, int n) {
  int i = blockIdx.x * blockDim.x + threadIdx.x;
  if (i < n) p[i] = 0.0f;
}

// fp32 W (Ksrc x N, row major) -> f16 B-fragment order, K zero-padded.
// Thread id == output f16 index, so stores are fully coalesced (one-time cost).
// B layout: lane c (=n%16) holds column n; lanes 0-15 K {0..7,16..23}, lanes 16-31 K {8..15,24..31}.
__global__ void snn_convert_w(const float* __restrict__ W, _Float16* __restrict__ Bfrag,
                              int Ktiles, int Ksrc, int N, int total) {
  int i = blockIdx.x * blockDim.x + threadIdx.x;
  if (i >= total) return;
  int pos  = i & 15;
  int lane = (i >> 4) & 31;
  int tile = i >> 9;                 // = nt * Ktiles + kt
  int kt = tile % Ktiles;
  int nt = tile / Ktiles;
  int k = kt * 32 + ((pos >> 3) << 4) + ((lane >> 4) << 3) + (pos & 7);
  int n = nt * 16 + (lane & 15);
  float v = (k < Ksrc) ? W[(size_t)k * N + n] : 0.0f;
  Bfrag[i] = (_Float16)v;
}

// xs = x / max(max(x), 1e-12) * SCALE   (single block: reduce then broadcast)
__global__ void snn_scale_x(const float* __restrict__ x, float* __restrict__ xs, int n) {
  __shared__ float red[256];
  float mx = 0.0f;
  for (int i = threadIdx.x; i < n; i += 256) mx = fmaxf(mx, x[i]);
  red[threadIdx.x] = mx;
  __syncthreads();
  for (int s = 128; s > 0; s >>= 1) {
    if (threadIdx.x < s) red[threadIdx.x] = fmaxf(red[threadIdx.x], red[threadIdx.x + s]);
    __syncthreads();
  }
  float sc = SCALE_F / fmaxf(red[0], 1e-12f);
  for (int i = threadIdx.x; i < n; i += 256) xs[i] = x[i] * sc;
}

// Layer-0 LIF: v = v*decay + xs; spike; write spike as f16 A-fragment for GEMM1.
__global__ void snn_lif0(const float* __restrict__ xs, float* __restrict__ v0,
                         float* __restrict__ sum0, _Float16* __restrict__ Afrag) {
  int i = blockIdx.x * blockDim.x + threadIdx.x;
  if (i >= IN_ELEMS) return;
  float v = v0[i] * DECAY_F + xs[i];
  bool s = (v >= THR_F);
  v0[i] = s ? 0.0f : v;
  if (s) sum0[i] += 1.0f;
  int m = i / INPUT_DIM, k = i % INPUT_DIM;
  Afrag[a_frag_off(m, k)] = s ? (_Float16)1.0f : (_Float16)0.0f;
}

// GEMM: C_part[seg] = A(64xK) @ B(Kx4096) over this block's K segment.
// 8 waves/block; wave -> one N-tile (16 cols) x 4 M-tiles (64 rows).
// The block's A-fragment slice (contiguous in fragment order) is DMA'd into
// LDS once by the Tensor Data Mover; all 8 waves consume it from LDS.
__global__ __launch_bounds__(256) void snn_gemm_wmma(
    const _Float16* __restrict__ Afrag, const _Float16* __restrict__ Bfrag,
    float* __restrict__ Cpart, int Ktiles, int ktPerSeg) {
  __shared__ __align__(128) _Float16 shA[SH_ELEMS];
  const int lane = threadIdx.x & 31;
  const int nt   = blockIdx.x * 8 + (threadIdx.x >> 5);
  const int seg  = blockIdx.y;
  const int kt0  = seg * ktPerSeg;

  if (threadIdx.x < 32) {  // wave 0 issues one TDM DMA for the whole block
    // shA is this kernel's only LDS object -> LDS byte offset 0
    tdm_load_1d(Afrag + (size_t)kt0 * FRAG_PER_KT, 0u,
                (unsigned)(ktPerSeg * FRAG_PER_KT));
    __builtin_amdgcn_s_wait_tensorcnt(0);
  }
  // The TDM intrinsic writes LDS via a descriptor the alias analysis cannot
  // see. Escape shA into a volatile asm with a memory clobber so the LDS
  // reads below are not folded to undef and stay ordered after the DMA.
  asm volatile("" : : "v"((const void*)shA) : "memory");
  __syncthreads();

  const v16h* AvL = (const v16h*)shA;
  const v16h* Bv  = (const v16h*)Bfrag;
  v8f acc0 = {}, acc1 = {}, acc2 = {}, acc3 = {};
  for (int kt = 0; kt < ktPerSeg; ++kt) {
    v16h b  = Bv[((size_t)nt * Ktiles + kt0 + kt) * 32 + lane];  // 32B contiguous/lane
    v16h a0 = AvL[(kt * 4 + 0) * 32 + lane];
    v16h a1 = AvL[(kt * 4 + 1) * 32 + lane];
    v16h a2 = AvL[(kt * 4 + 2) * 32 + lane];
    v16h a3 = AvL[(kt * 4 + 3) * 32 + lane];
    acc0 = __builtin_amdgcn_wmma_f32_16x16x32_f16(false, a0, false, b, (short)0, acc0, false, false);
    acc1 = __builtin_amdgcn_wmma_f32_16x16x32_f16(false, a1, false, b, (short)0, acc1, false, false);
    acc2 = __builtin_amdgcn_wmma_f32_16x16x32_f16(false, a2, false, b, (short)0, acc2, false, false);
    acc3 = __builtin_amdgcn_wmma_f32_16x16x32_f16(false, a3, false, b, (short)0, acc3, false, false);
  }
  // C/D layout: VGPR j -> row j (lanes 0-15) / row j+8 (lanes 16-31); col = lane%16
  float* Cs = Cpart + (size_t)seg * HID_ELEMS;
  const int n = nt * 16 + (lane & 15);
  const int rb = (lane >> 4) * 8;
#pragma unroll
  for (int j = 0; j < 8; ++j) {
    Cs[(size_t)(rb + j +  0) * HIDDEN + n] = acc0[j];
    Cs[(size_t)(rb + j + 16) * HIDDEN + n] = acc1[j];
    Cs[(size_t)(rb + j + 32) * HIDDEN + n] = acc2[j];
    Cs[(size_t)(rb + j + 48) * HIDDEN + n] = acc3[j];
  }
}

// Hidden-layer LIF: reduce K-split partials, update v, accumulate spike sum
// (kwta on binary spikes is the identity), optionally emit next A-fragments.
__global__ void snn_lif_hidden(const float* __restrict__ Cpart, int nseg,
                               float* __restrict__ v, float* __restrict__ sum,
                               _Float16* __restrict__ Afrag) {
  int i = blockIdx.x * blockDim.x + threadIdx.x;
  if (i >= HID_ELEMS) return;
  float cur = 0.0f;
  for (int s = 0; s < nseg; ++s) cur += Cpart[(size_t)s * HID_ELEMS + i];
  float vv = v[i] * DECAY_F + cur;
  bool sp = (vv >= THR_F);
  v[i] = sp ? 0.0f : vv;
  if (sp) sum[i] += 1.0f;
  if (Afrag) {
    int m = i >> 12, k = i & (HIDDEN - 1);
    Afrag[a_frag_off(m, k)] = sp ? (_Float16)1.0f : (_Float16)0.0f;
  }
}

__global__ void snn_finalize(const float* __restrict__ s0, const float* __restrict__ s1,
                             const float* __restrict__ s2, float* __restrict__ out) {
  int i = blockIdx.x * blockDim.x + threadIdx.x;
  const float inv = 1.0f / (float)T_STEPS;
  if (i < IN_ELEMS)                          out[i] = s0[i] * inv;
  else if (i < IN_ELEMS + HID_ELEMS)         out[i] = s1[i - IN_ELEMS] * inv;
  else if (i < IN_ELEMS + 2 * HID_ELEMS)     out[i] = s2[i - IN_ELEMS - HID_ELEMS] * inv;
}

// ---- launch ---------------------------------------------------------------

extern "C" void kernel_launch(void* const* d_in, const int* in_sizes, int n_in,
                              void* d_out, int out_size, void* d_ws, size_t ws_size,
                              hipStream_t stream) {
  const float* x  = (const float*)d_in[0];   // 64 x 784
  const float* W1 = (const float*)d_in[1];   // 784 x 4096
  const float* W2 = (const float*)d_in[2];   // 4096 x 4096
  float* out = (float*)d_out;

  char* p = (char*)d_ws;
  auto carve = [&](size_t bytes) { void* r = (void*)p; p += (bytes + 255) & ~(size_t)255; return r; };
  _Float16* W1h   = (_Float16*)carve((size_t)K1P * HIDDEN * 2);      // 6.25 MB
  _Float16* W2h   = (_Float16*)carve((size_t)HIDDEN * HIDDEN * 2);   // 32 MB
  _Float16* A1    = (_Float16*)carve((size_t)BATCH * K1P * 2);       // s0 frags
  _Float16* A2    = (_Float16*)carve((size_t)BATCH * HIDDEN * 2);    // s1 frags
  float* xs    = (float*)carve((size_t)IN_ELEMS * 4);
  float* v0    = (float*)carve((size_t)IN_ELEMS * 4);
  float* v1    = (float*)carve((size_t)HID_ELEMS * 4);
  float* v2    = (float*)carve((size_t)HID_ELEMS * 4);
  float* sum0  = (float*)carve((size_t)IN_ELEMS * 4);
  float* sum1  = (float*)carve((size_t)HID_ELEMS * 4);
  float* sum2  = (float*)carve((size_t)HID_ELEMS * 4);
  float* cur1p = (float*)carve((size_t)KSEG1 * HID_ELEMS * 4);
  float* cur2p = (float*)carve((size_t)KSEG2 * HID_ELEMS * 4);

  const int TB = 256;
  auto blocks = [](int n) { return (n + 255) / 256; };

  // One-time: zero state + sums + A1 (covers K padding region)
  snn_zero_f32<<<blocks(IN_ELEMS),  TB, 0, stream>>>(v0,   IN_ELEMS);
  snn_zero_f32<<<blocks(HID_ELEMS), TB, 0, stream>>>(v1,   HID_ELEMS);
  snn_zero_f32<<<blocks(HID_ELEMS), TB, 0, stream>>>(v2,   HID_ELEMS);
  snn_zero_f32<<<blocks(IN_ELEMS),  TB, 0, stream>>>(sum0, IN_ELEMS);
  snn_zero_f32<<<blocks(HID_ELEMS), TB, 0, stream>>>(sum1, HID_ELEMS);
  snn_zero_f32<<<blocks(HID_ELEMS), TB, 0, stream>>>(sum2, HID_ELEMS);
  snn_zero_f32<<<blocks(BATCH * K1P / 2), TB, 0, stream>>>((float*)A1, BATCH * K1P / 2);

  // One-time: weight conversion to f16 B-fragment order; input scaling
  const int totW1 = K1P * HIDDEN;
  const int totW2 = HIDDEN * HIDDEN;
  snn_convert_w<<<blocks(totW1), TB, 0, stream>>>(W1, W1h, KT1, INPUT_DIM, HIDDEN, totW1);
  snn_convert_w<<<blocks(totW2), TB, 0, stream>>>(W2, W2h, KT2, HIDDEN,    HIDDEN, totW2);
  snn_scale_x<<<1, TB, 0, stream>>>(x, xs, IN_ELEMS);

  // Sequential time loop (state carried through v0/v1/v2)
  for (int t = 0; t < T_STEPS; ++t) {
    snn_lif0<<<blocks(IN_ELEMS), TB, 0, stream>>>(xs, v0, sum0, A1);
    snn_gemm_wmma<<<dim3(HIDDEN / 128, KSEG1), TB, 0, stream>>>(A1, W1h, cur1p, KT1, KT1 / KSEG1);
    snn_lif_hidden<<<blocks(HID_ELEMS), TB, 0, stream>>>(cur1p, KSEG1, v1, sum1, A2);
    snn_gemm_wmma<<<dim3(HIDDEN / 128, KSEG2), TB, 0, stream>>>(A2, W2h, cur2p, KT2, KT2 / KSEG2);
    snn_lif_hidden<<<blocks(HID_ELEMS), TB, 0, stream>>>(cur2p, KSEG2, v2, sum2, nullptr);
  }

  snn_finalize<<<blocks(IN_ELEMS + 2 * HID_ELEMS), TB, 0, stream>>>(sum0, sum1, sum2, out);
}
